// Model_48026324304037
// MI455X (gfx1250) — matrix operand order repaired and checked by reading, not compiled, not run on hardware
//
#include <hip/hip_runtime.h>
#include <hip/hip_bf16.h>
#include <math.h>

typedef __attribute__((ext_vector_type(16))) _Float16 v16h;
typedef __attribute__((ext_vector_type(8)))  float    v8f;

union AFrag { v16h v; _Float16 e[16]; };
union CAcc  { v8f  v; float    e[8];  };

static constexpr int Bb = 16;        // batch
static constexpr int Ss = 256;       // seq len
static constexpr int Kw = 16;        // window / timesteps
static constexpr int SK = Ss + Kw;   // 272 padded positions per batch row
static constexpr int Ee = 512;       // embed dim
static constexpr int Hh = 512;       // hidden dim
static constexpr int G3 = 3 * Hh;    // 1536 gate columns
static constexpr int Nn = Bb * Ss;   // 4096 independent sequences
static constexpr int Tt = Kw;        // 16 GRU steps
static constexpr int Mr0 = 64;       // rows per block, GRU layer 0 (4 row tiles)
static constexpr int Mr1 = 32;       // rows per block, GRU layer 1 (2 row tiles)

// A-matrix 16x32 f16 fragment (ISA 7.12.2): lanes 0-15 row M=lane, elems 0..7 = K 0..7,
// elems 8..15 = K 16..23; lanes 16-31 same rows, K 8..15 / 24..31.
__device__ __forceinline__ v16h load_fragA(const _Float16* row, int k0, int hf) {
  AFrag f;
  const _Float16* p0 = row + k0 + hf * 8;
  const _Float16* p1 = row + k0 + 16 + hf * 8;
#pragma unroll
  for (int i = 0; i < 8; ++i) { f.e[i] = p0[i]; f.e[8 + i] = p1[i]; }
  return f.v;
}

// B-matrix 32x16 f16 fragment: lane holds column N=lane%16; lanes 0-15 elems 0..7 = K 0..7,
// elems 8..15 = K 16..23; lanes 16-31 K 8..15 / 24..31. B[k][n] = W[n][k] (row-major W, k contiguous).
__device__ __forceinline__ v16h load_fragB(const _Float16* wrow, int k0, int hf) {
  AFrag f;
  const _Float16* p0 = wrow + k0 + hf * 8;
  const _Float16* p1 = wrow + k0 + 16 + hf * 8;
#pragma unroll
  for (int i = 0; i < 8; ++i) { f.e[i] = p0[i]; f.e[8 + i] = p1[i]; }
  return f.v;
}

__device__ __forceinline__ float fast_rcp(float x) {
#if __has_builtin(__builtin_amdgcn_rcpf)
  return __builtin_amdgcn_rcpf(x);
#else
  return 1.0f / x;
#endif
}
// sigmoid(x) = rcp(1 + exp(-x))         -> v_mul, v_exp, v_add, v_rcp
__device__ __forceinline__ float sigmoid_fast(float x) {
  return fast_rcp(1.0f + __expf(-x));
}
// tanh(x) = 2*rcp(1 + exp(-2x)) - 1     -> v_mul, v_exp, v_add, v_rcp, v_fma
__device__ __forceinline__ float tanh_fast(float x) {
  return __builtin_fmaf(2.0f, fast_rcp(1.0f + __expf(-2.0f * x)), -1.0f);
}

// ---------------- prep kernels ----------------

__global__ void cvt_f16_kernel(const float* __restrict__ src, _Float16* __restrict__ dst, int n) {
  int i = blockIdx.x * blockDim.x + threadIdx.x;
  if (i < n) dst[i] = (_Float16)src[i];
}

__global__ void zero_f32_kernel(float* __restrict__ p, int n) {
  int i = blockIdx.x * blockDim.x + threadIdx.x;
  if (i < n) p[i] = 0.0f;
}

// Ef[b*SK + p][e] = f16(embd[token(b,p)][e]); token = 0 pad for p < Kw (embd row 0 is zero)
__global__ void gather_embed_kernel(const int* __restrict__ x, const float* __restrict__ embd,
                                    _Float16* __restrict__ Ef) {
  const int row = blockIdx.x;                // 0 .. Bb*SK-1
  const int b = row / SK, p = row % SK;
  const int tok = (p < Kw) ? 0 : x[b * Ss + (p - Kw)];
  const float* src = embd + (size_t)tok * Ee;
  _Float16* dst = Ef + (size_t)row * Ee;
  for (int e = threadIdx.x; e < Ee; e += blockDim.x) dst[e] = (_Float16)src[e];
}

// ---------------- P0 = Ef @ w_ih0^T + b_ih0  (4352 x 1536, f32) ----------------

__global__ __launch_bounds__(256) void proj0_kernel(
    const _Float16* __restrict__ Ef, const _Float16* __restrict__ wih,
    const float* __restrict__ bih, float* __restrict__ P0) {
  const int lane = threadIdx.x & 31;
  const int wave = threadIdx.x >> 5;
  const int hf = lane >> 4;
  const int lr = lane & 15;
  const int r0 = blockIdx.x * 16;
  const int c  = (blockIdx.y * 8 + wave) * 16 + lr;   // output col / W row
  CAcc acc;
#pragma unroll
  for (int v = 0; v < 8; ++v) acc.e[v] = 0.0f;
  for (int kt = 0; kt < Ee / 32; ++kt) {
    const int k0 = kt * 32;
    v16h a  = load_fragA(Ef  + (size_t)(r0 + lr) * Ee, k0, hf);
    v16h bf = load_fragB(wih + (size_t)c * Ee,        k0, hf);
    acc.v = __builtin_amdgcn_wmma_f32_16x16x32_f16(false, a, false, bf, (short)0, acc.v,
                                                   false, false);
  }
  const float bias = bih[c];
#pragma unroll
  for (int v = 0; v < 8; ++v)
    P0[(size_t)(r0 + v + 8 * hf) * G3 + c] = acc.e[v] + bias;
}

// ---------------- GRU layer 0: per-block persistent recurrence ----------------
// 512-thread block owns Mr0=64 rows (same b). Wave w owns 32 H-cols [w*32, w*32+32)
// and the matching 3 gate col-tile pairs; each B fragment is reused across the 4 row
// tiles (4x register-level B reuse). h lives in LDS (f16); gates run on WMMA accs.
// Acc tiles/wave: 4 rt * 3 g * 2 ct = 24 (192 VGPRs).

__global__ __launch_bounds__(512) void gru0_kernel(
    const _Float16* __restrict__ whh, const float* __restrict__ bhh,
    const float* __restrict__ P0, _Float16* __restrict__ out0) {
  __shared__ _Float16 hbuf[Mr0 * Hh];
  const int lane = threadIdx.x & 31;
  const int wave = threadIdx.x >> 5;     // 0..15
  const int hf = lane >> 4;
  const int lr = lane & 15;
  const int n0 = blockIdx.x * Mr0;
  const int b  = n0 / Ss;
  const int s0 = n0 % Ss;
  const int jbase = wave * 32;

  for (int i = threadIdx.x; i < Mr0 * Hh; i += 512) hbuf[i] = (_Float16)0.0f;
  __syncthreads();

  for (int t = 0; t < Tt; ++t) {
    CAcc acc[24];                          // [rt*6 + g*2 + ct]
#pragma unroll
    for (int i = 0; i < 24; ++i)
#pragma unroll
      for (int v = 0; v < 8; ++v) acc[i].e[v] = 0.0f;

    for (int kt = 0; kt < Hh / 32; ++kt) {
      const int k0 = kt * 32;
      v16h a[4];
#pragma unroll
      for (int rt = 0; rt < 4; ++rt)
        a[rt] = load_fragA(&hbuf[(rt * 16 + lr) * Hh], k0, hf);
#pragma unroll
      for (int g = 0; g < 3; ++g) {
#pragma unroll
        for (int ct = 0; ct < 2; ++ct) {
          const int col = g * Hh + jbase + ct * 16 + lr;
          v16h bf = load_fragB(whh + (size_t)col * Hh, k0, hf);
#pragma unroll
          for (int rt = 0; rt < 4; ++rt) {
            acc[rt * 6 + g * 2 + ct].v = __builtin_amdgcn_wmma_f32_16x16x32_f16(
                false, a[rt], false, bf, (short)0, acc[rt * 6 + g * 2 + ct].v, false, false);
          }
        }
      }
    }
    __syncthreads();   // all waves done reading h before owners rewrite it

#pragma unroll
    for (int rt = 0; rt < 4; ++rt) {
#pragma unroll
      for (int ct = 0; ct < 2; ++ct) {
        const int j = jbase + ct * 16 + lr;
        const float bhr = bhh[j], bhz = bhh[Hh + j], bhn = bhh[2 * Hh + j];
#pragma unroll
        for (int v = 0; v < 8; ++v) {
          const int row = rt * 16 + v + 8 * hf;
          const int s = s0 + row;
          const float* pr = P0 + (size_t)(b * SK + s + t) * G3;  // dedup'd x_proj (b_ih folded)
          const float r  = sigmoid_fast(pr[j]      + acc[rt * 6 + ct].e[v]     + bhr);
          const float z  = sigmoid_fast(pr[Hh + j] + acc[rt * 6 + 2 + ct].e[v] + bhz);
          const float nn = tanh_fast(pr[2 * Hh + j] + r * (acc[rt * 6 + 4 + ct].e[v] + bhn));
          const float hp = (float)hbuf[row * Hh + j];
          const float hnew = (1.0f - z) * nn + z * hp;
          hbuf[row * Hh + j] = (_Float16)hnew;
          out0[((size_t)t * Nn + (n0 + row)) * Hh + j] = (_Float16)hnew;
        }
      }
    }
    __syncthreads();
  }
}

// ---------------- GRU layer 1: fused input-proj + recurrence; only h_last kept --------
// 512-thread block, Mr1=32 rows, 32 cols/wave; acc tiles/wave:
// 2 rt * 3 g * 2 ct * 2 matrices = 24 (192 VGPRs). Each B frag feeds 2 WMMAs.

__global__ __launch_bounds__(512) void gru1_kernel(
    const _Float16* __restrict__ wih, const _Float16* __restrict__ whh,
    const float* __restrict__ bih, const float* __restrict__ bhh,
    const _Float16* __restrict__ out0, _Float16* __restrict__ hlast) {
  __shared__ _Float16 hbuf[Mr1 * Hh];
  const int lane = threadIdx.x & 31;
  const int wave = threadIdx.x >> 5;     // 0..15
  const int hf = lane >> 4;
  const int lr = lane & 15;
  const int n0 = blockIdx.x * Mr1;
  const int jbase = wave * 32;

  for (int i = threadIdx.x; i < Mr1 * Hh; i += 512) hbuf[i] = (_Float16)0.0f;
  __syncthreads();

  for (int t = 0; t < Tt; ++t) {
    CAcc ax[12], ah[12];                   // [rt*6 + g*2 + ct]
#pragma unroll
    for (int i = 0; i < 12; ++i)
#pragma unroll
      for (int v = 0; v < 8; ++v) { ax[i].e[v] = 0.0f; ah[i].e[v] = 0.0f; }

    for (int kt = 0; kt < Hh / 32; ++kt) {
      const int k0 = kt * 32;
      v16h aX0 = load_fragA(out0 + ((size_t)t * Nn + n0 + lr) * Hh,      k0, hf);
      v16h aX1 = load_fragA(out0 + ((size_t)t * Nn + n0 + 16 + lr) * Hh, k0, hf);
      v16h aH0 = load_fragA(&hbuf[lr * Hh],        k0, hf);
      v16h aH1 = load_fragA(&hbuf[(16 + lr) * Hh], k0, hf);
#pragma unroll
      for (int g = 0; g < 3; ++g) {
#pragma unroll
        for (int ct = 0; ct < 2; ++ct) {
          const int col = g * Hh + jbase + ct * 16 + lr;
          const int ix = g * 2 + ct;
          v16h bi = load_fragB(wih + (size_t)col * Hh, k0, hf);
          ax[ix].v = __builtin_amdgcn_wmma_f32_16x16x32_f16(
              false, aX0, false, bi, (short)0, ax[ix].v, false, false);
          ax[6 + ix].v = __builtin_amdgcn_wmma_f32_16x16x32_f16(
              false, aX1, false, bi, (short)0, ax[6 + ix].v, false, false);
          v16h bh = load_fragB(whh + (size_t)col * Hh, k0, hf);
          ah[ix].v = __builtin_amdgcn_wmma_f32_16x16x32_f16(
              false, aH0, false, bh, (short)0, ah[ix].v, false, false);
          ah[6 + ix].v = __builtin_amdgcn_wmma_f32_16x16x32_f16(
              false, aH1, false, bh, (short)0, ah[6 + ix].v, false, false);
        }
      }
    }
    __syncthreads();

#pragma unroll
    for (int rt = 0; rt < 2; ++rt) {
#pragma unroll
      for (int ct = 0; ct < 2; ++ct) {
        const int j = jbase + ct * 16 + lr;
        const float bir = bih[j], biz = bih[Hh + j], bin = bih[2 * Hh + j];
        const float bhr = bhh[j], bhz = bhh[Hh + j], bhn = bhh[2 * Hh + j];
#pragma unroll
        for (int v = 0; v < 8; ++v) {
          const int row = rt * 16 + v + 8 * hf;
          const int o = rt * 6 + ct;
          const float r  = sigmoid_fast((ax[o].e[v] + bir)     + (ah[o].e[v] + bhr));
          const float z  = sigmoid_fast((ax[o + 2].e[v] + biz) + (ah[o + 2].e[v] + bhz));
          const float nn = tanh_fast((ax[o + 4].e[v] + bin) + r * (ah[o + 4].e[v] + bhn));
          const float hp = (float)hbuf[row * Hh + j];
          const float hnew = (1.0f - z) * nn + z * hp;
          hbuf[row * Hh + j] = (_Float16)hnew;
          if (t == Tt - 1) hlast[(size_t)(n0 + row) * Hh + j] = (_Float16)hnew;
        }
      }
    }
    __syncthreads();
  }
}

// ---------------- MLP + ReLU + max-pool over S (atomicMax on non-negative f32) --------

__global__ __launch_bounds__(256) void mlp_pool_kernel(
    const _Float16* __restrict__ hlast, const _Float16* __restrict__ mlpw,
    const float* __restrict__ mlpb, float* __restrict__ pooled) {
  const int lane = threadIdx.x & 31;
  const int wave = threadIdx.x >> 5;
  const int hf = lane >> 4;
  const int lr = lane & 15;
  const int n0 = blockIdx.x * 16;
  const int b = n0 / Ss;
  const int c = (blockIdx.y * 8 + wave) * 16 + lr;
  CAcc acc;
#pragma unroll
  for (int v = 0; v < 8; ++v) acc.e[v] = 0.0f;
  for (int kt = 0; kt < Hh / 32; ++kt) {
    const int k0 = kt * 32;
    v16h a  = load_fragA(hlast + (size_t)(n0 + lr) * Hh, k0, hf);
    v16h bf = load_fragB(mlpw + (size_t)c * Hh, k0, hf);
    acc.v = __builtin_amdgcn_wmma_f32_16x16x32_f16(false, a, false, bf, (short)0, acc.v,
                                                   false, false);
  }
  const float bias = mlpb[c];
  float m = 0.0f;  // ReLU outputs are >= 0, so 0 is the max identity here
#pragma unroll
  for (int v = 0; v < 8; ++v) m = fmaxf(m, fmaxf(acc.e[v] + bias, 0.0f));
  m = fmaxf(m, __shfl_xor(m, 16, 32));           // combine the two row-halves (same col)
  if (hf == 0)
    atomicMax((int*)&pooled[(size_t)b * Hh + c], __float_as_int(m));  // ok: m >= 0
}

// ---------------- tiny decode: out[b][c] = pooled[b] . dec_w[c] + dec_b[c] ------------

__global__ void decode_kernel(const float* __restrict__ pooled, const float* __restrict__ decw,
                              const float* __restrict__ decb, float* __restrict__ out) {
  const int tid = threadIdx.x;
  if (tid >= Bb * 2) return;
  const int b = tid >> 1, c = tid & 1;
  float s = decb[c];
  for (int k = 0; k < Hh; ++k) s += pooled[(size_t)b * Hh + k] * decw[(size_t)c * Hh + k];
  out[b * 2 + c] = s;
}

// ---------------- launch ----------------

extern "C" void kernel_launch(void* const* d_in, const int* in_sizes, int n_in,
                              void* d_out, int out_size, void* d_ws, size_t ws_size,
                              hipStream_t stream) {
  (void)in_sizes; (void)n_in; (void)out_size; (void)ws_size;
  const int*   x     = (const int*)d_in[0];
  // d_in[1] = lens (unused by the reference)
  const float* embd  = (const float*)d_in[2];
  const float* w_ih0 = (const float*)d_in[3];
  const float* w_hh0 = (const float*)d_in[4];
  const float* b_ih0 = (const float*)d_in[5];
  const float* b_hh0 = (const float*)d_in[6];
  const float* w_ih1 = (const float*)d_in[7];
  const float* w_hh1 = (const float*)d_in[8];
  const float* b_ih1 = (const float*)d_in[9];
  const float* b_hh1 = (const float*)d_in[10];
  const float* mlp_w = (const float*)d_in[11];
  const float* mlp_b = (const float*)d_in[12];
  const float* dec_w = (const float*)d_in[13];
  const float* dec_b = (const float*)d_in[14];
  float* out = (float*)d_out;

  char* base = (char*)d_ws;
  size_t off = 0;
  auto take = [&](size_t bytes) -> void* {
    void* r = base + off;
    off += (bytes + 255) & ~(size_t)255;
    return r;
  };
  _Float16* wih0h = (_Float16*)take((size_t)G3 * Ee * 2);
  _Float16* whh0h = (_Float16*)take((size_t)G3 * Hh * 2);
  _Float16* wih1h = (_Float16*)take((size_t)G3 * Hh * 2);
  _Float16* whh1h = (_Float16*)take((size_t)G3 * Hh * 2);
  _Float16* mlph  = (_Float16*)take((size_t)Hh * Hh * 2);
  _Float16* Ef    = (_Float16*)take((size_t)Bb * SK * Ee * 2);
  float*    P0    = (float*)   take((size_t)Bb * SK * G3 * 4);
  _Float16* out0  = (_Float16*)take((size_t)Tt * Nn * Hh * 2);
  _Float16* hlast = (_Float16*)take((size_t)Nn * Hh * 2);
  float*    pooled= (float*)   take((size_t)Bb * Hh * 4);

  const int thr = 256;
  const int ng = G3 * Ee;   // 786432
  cvt_f16_kernel<<<(ng + thr - 1) / thr, thr, 0, stream>>>(w_ih0, wih0h, ng);
  cvt_f16_kernel<<<(ng + thr - 1) / thr, thr, 0, stream>>>(w_hh0, whh0h, ng);
  cvt_f16_kernel<<<(ng + thr - 1) / thr, thr, 0, stream>>>(w_ih1, wih1h, ng);
  cvt_f16_kernel<<<(ng + thr - 1) / thr, thr, 0, stream>>>(w_hh1, whh1h, ng);
  const int nm = Hh * Hh;
  cvt_f16_kernel<<<(nm + thr - 1) / thr, thr, 0, stream>>>(mlp_w, mlph, nm);

  gather_embed_kernel<<<Bb * SK, thr, 0, stream>>>(x, embd, Ef);
  proj0_kernel<<<dim3(Bb * SK / 16, G3 / (8 * 16)), thr, 0, stream>>>(Ef, wih0h, b_ih0, P0);
  gru0_kernel<<<Nn / Mr0, 512, 0, stream>>>(whh0h, b_hh0, P0, out0);
  gru1_kernel<<<Nn / Mr1, 512, 0, stream>>>(wih1h, whh1h, b_ih1, b_hh1, out0, hlast);
  zero_f32_kernel<<<(Bb * Hh + thr - 1) / thr, thr, 0, stream>>>(pooled, Bb * Hh);
  mlp_pool_kernel<<<dim3(Nn / 16, Hh / (8 * 16)), thr, 0, stream>>>(hlast, mlph, mlp_b, pooled);
  decode_kernel<<<1, 32, 0, stream>>>(pooled, dec_w, dec_b, out);
}
